// Attention_13984413516503
// MI455X (gfx1250) — matrix operand order repaired and checked by reading, not compiled, not run on hardware
//
#include <hip/hip_runtime.h>
#include <math.h>

#define DEV __device__ __forceinline__

typedef __attribute__((ext_vector_type(16))) __bf16 v16bf;
typedef __attribute__((ext_vector_type(8)))  float  v8f;
typedef int v4i_ __attribute__((ext_vector_type(4)));

union BFrag { v16bf v; uint4 q[2]; };
union CFrag { v8f v; float f[8]; };

DEV unsigned short f2bf(float f) {
    unsigned u = __float_as_uint(f);
    u += 0x7FFFu + ((u >> 16) & 1u);   // round-to-nearest-even
    return (unsigned short)(u >> 16);
}
DEV float bf2f(unsigned short h) { return __uint_as_float(((unsigned)h) << 16); }

// ---- CDNA5 async global->LDS path (guarded; fallback = sync staging) ------
#if __has_builtin(__builtin_amdgcn_global_load_async_to_lds_b128)
#define HAS_ASYNC_LDS 1
#define ASYNC_CP16(gsrc, ldst)                                        \
    __builtin_amdgcn_global_load_async_to_lds_b128(                   \
        (v4i_*)(gsrc), (v4i_*)(ldst), 0, 0)
#else
#define HAS_ASYNC_LDS 0
#endif

#if __has_builtin(__builtin_amdgcn_s_wait_asynccnt)
#define WAIT_ASYNC(n) __builtin_amdgcn_s_wait_asynccnt(n)
#else
#define WAIT_ASYNC(n) asm volatile("s_wait_asynccnt %0" ::"i"(n) : "memory")
#endif

// ---------------------------------------------------------------------------
// fp32 -> bf16 cast
// ---------------------------------------------------------------------------
__global__ void cast_f32_bf16(const float* __restrict__ in,
                              unsigned short* __restrict__ out, int n) {
    int i = blockIdx.x * 256 + threadIdx.x;
    if (i < n) out[i] = f2bf(in[i]);
}

// ---------------------------------------------------------------------------
// C[MxN] (f32) = A[MxK] (bf16, row-major) @ B[NxK]^T (bf16, row-major)
// 128x128 block tile, K-step 32, 8 waves, each wave: 2x4 grid of 16x16 WMMAs.
// Double-buffered LDS tiles filled by GLOBAL_LOAD_ASYNC_TO_LDS_B128.
// ---------------------------------------------------------------------------
__global__ __launch_bounds__(256) void gemm_bf16_nt(
    const unsigned short* __restrict__ A,
    const unsigned short* __restrict__ B,
    float* __restrict__ C, int M, int N, int K) {
    __shared__ unsigned short As[2][128][40];  // 80B rows (16B aligned)
    __shared__ unsigned short Bs[2][128][40];

    const int m0 = blockIdx.y * 128;
    const int n0 = blockIdx.x * 128;
    const int tid = threadIdx.x;
    const int wave = tid >> 5, lane = tid & 31;
    const int wr = wave & 3;   // row group  (32 rows)
    const int wc = wave >> 2;  // col group  (64 cols)

    CFrag acc[2][4];
#pragma unroll
    for (int i = 0; i < 2; ++i)
#pragma unroll
        for (int j = 0; j < 4; ++j)
#pragma unroll
            for (int e = 0; e < 8; ++e) acc[i][j].f[e] = 0.f;

    const int lr = tid >> 1;         // 0..127 : tile row to load
    const int lc = (tid & 1) * 16;   // 0 / 16 : half-offset in K
    // clamp rows: OOB garbage only reaches C-rows >= M, which are never stored
    const int rA = (m0 + lr < M) ? (m0 + lr) : (M - 1);
    const int rB = (n0 + lr < N) ? (n0 + lr) : (N - 1);
    const unsigned short* gA = A + (size_t)rA * K + lc;
    const unsigned short* gB = B + (size_t)rB * K + lc;

    const int nk = K >> 5;

#if HAS_ASYNC_LDS
    // prologue: stream tile 0 into buffer 0 (4 async b128 ops per thread)
    ASYNC_CP16(gA,     &As[0][lr][lc]);
    ASYNC_CP16(gA + 8, &As[0][lr][lc + 8]);
    ASYNC_CP16(gB,     &Bs[0][lr][lc]);
    ASYNC_CP16(gB + 8, &Bs[0][lr][lc + 8]);

    for (int i = 0; i < nk; ++i) {
        const int cur = i & 1;
        if (i + 1 < nk) {
            const int k = (i + 1) << 5;
            const int nxt = cur ^ 1;
            ASYNC_CP16(gA + k,     &As[nxt][lr][lc]);
            ASYNC_CP16(gA + k + 8, &As[nxt][lr][lc + 8]);
            ASYNC_CP16(gB + k,     &Bs[nxt][lr][lc]);
            ASYNC_CP16(gB + k + 8, &Bs[nxt][lr][lc + 8]);
            WAIT_ASYNC(4);  // in-order: current buffer's 4 ops are done
        } else {
            WAIT_ASYNC(0);
        }
        __syncthreads();

        BFrag af[2], bfr[4];
#pragma unroll
        for (int ii = 0; ii < 2; ++ii) {            // A frag: ISA 16-bit A layout
            int r = wr * 32 + ii * 16 + (lane & 15);
            int kb = (lane < 16) ? 0 : 8;
            af[ii].q[0] = *(const uint4*)&As[cur][r][kb];
            af[ii].q[1] = *(const uint4*)&As[cur][r][kb + 16];
        }
#pragma unroll
        for (int j = 0; j < 4; ++j) {               // B frag: ISA 16-bit layout (same as A)
            int c = wc * 64 + j * 16 + (lane & 15);
            int kb = (lane < 16) ? 0 : 8;
            bfr[j].q[0] = *(const uint4*)&Bs[cur][c][kb];
            bfr[j].q[1] = *(const uint4*)&Bs[cur][c][kb + 16];
        }
#pragma unroll
        for (int ii = 0; ii < 2; ++ii)
#pragma unroll
            for (int j = 0; j < 4; ++j)
                acc[ii][j].v = __builtin_amdgcn_wmma_f32_16x16x32_bf16(
                    false, af[ii].v, false, bfr[j].v, (short)0, acc[ii][j].v,
                    false, false);
        __syncthreads();  // all waves done with buf[cur] before it is refilled
    }
#else
    // fallback: synchronous staging through VGPRs, single buffer
    for (int i = 0; i < nk; ++i) {
        const int k0 = i << 5;
        const uint4* pa = (const uint4*)(gA + k0);
        const uint4* pb = (const uint4*)(gB + k0);
        uint4 a0 = pa[0], a1 = pa[1], b0 = pb[0], b1 = pb[1];
        *(uint4*)&As[0][lr][lc] = a0; *(uint4*)&As[0][lr][lc + 8] = a1;
        *(uint4*)&Bs[0][lr][lc] = b0; *(uint4*)&Bs[0][lr][lc + 8] = b1;
        __syncthreads();
        BFrag af[2], bfr[4];
#pragma unroll
        for (int ii = 0; ii < 2; ++ii) {
            int r = wr * 32 + ii * 16 + (lane & 15);
            int kb = (lane < 16) ? 0 : 8;
            af[ii].q[0] = *(const uint4*)&As[0][r][kb];
            af[ii].q[1] = *(const uint4*)&As[0][r][kb + 16];
        }
#pragma unroll
        for (int j = 0; j < 4; ++j) {
            int c = wc * 64 + j * 16 + (lane & 15);
            int kb = (lane < 16) ? 0 : 8;
            bfr[j].q[0] = *(const uint4*)&Bs[0][c][kb];
            bfr[j].q[1] = *(const uint4*)&Bs[0][c][kb + 16];
        }
#pragma unroll
        for (int ii = 0; ii < 2; ++ii)
#pragma unroll
            for (int j = 0; j < 4; ++j)
                acc[ii][j].v = __builtin_amdgcn_wmma_f32_16x16x32_bf16(
                    false, af[ii].v, false, bfr[j].v, (short)0, acc[ii][j].v,
                    false, false);
        __syncthreads();
    }
#endif

    const int rowsel = (lane >> 4) * 8;
    const int nc = lane & 15;
#pragma unroll
    for (int i = 0; i < 2; ++i)
#pragma unroll
        for (int j = 0; j < 4; ++j) {
            int mb = m0 + wr * 32 + i * 16 + rowsel;
            int nb = n0 + wc * 64 + j * 16 + nc;
#pragma unroll
            for (int e = 0; e < 8; ++e) {
                int m = mb + e;
                if (m < M && nb < N) C[(size_t)m * N + nb] = acc[i][j].f[e];
            }
        }
}

// ---------------------------------------------------------------------------
// RoPE + repack [b*s][2048] f32 -> [b][h][s][128] bf16
// ---------------------------------------------------------------------------
__global__ void rope_pack(const float* __restrict__ T,
                          const float* __restrict__ Cos,
                          const float* __restrict__ Sin,
                          unsigned short* __restrict__ out) {
    int bs = blockIdx.x;               // b*1024 + s
    int s = bs & 1023, b = bs >> 10;
    int tid = threadIdx.x;             // 0..1023
    int h = tid >> 6, j = tid & 63;
    size_t src = (size_t)bs * 2048 + h * 128 + 2 * j;
    float xr = T[src], xi = T[src + 1];
    float c = Cos[s * 64 + j], sn = Sin[s * 64 + j];
    size_t o = ((size_t)(b * 16 + h) * 1024 + s) * 128 + 2 * j;
    out[o]     = f2bf(xr * c - xi * sn);
    out[o + 1] = f2bf(xr * sn + xi * c);
}

// ---------------------------------------------------------------------------
// V repack + transpose: [b*s][2048] f32 -> [b][h][128][1024] bf16
// ---------------------------------------------------------------------------
__global__ void pack_vt(const float* __restrict__ V,
                        unsigned short* __restrict__ out) {
    int bs = blockIdx.x; int s = bs & 1023, b = bs >> 10;
    for (int c = threadIdx.x; c < 2048; c += 256) {
        int h = c >> 7, d = c & 127;
        out[((size_t)(b * 16 + h) * 128 + d) * 1024 + s] =
            f2bf(V[(size_t)bs * 2048 + c]);
    }
}

// ---------------------------------------------------------------------------
// Flash attention per (b, h, 64-row q tile). Causal + gate2 window + separate
// 10-key adapter softmax * tanh(gate1). Output bf16 [b*s][2048].
// ---------------------------------------------------------------------------
__global__ __launch_bounds__(256) void attn_flash(
    const unsigned short* __restrict__ Q,   // [b][h][1024][128]
    const unsigned short* __restrict__ Kc,  // [b][h][1024][128]
    const unsigned short* __restrict__ VT,  // [b][h][128][1024]
    const float* __restrict__ AK,           // [10][2048]
    const float* __restrict__ AV,           // [10][2048]
    const float* __restrict__ g1, const float* __restrict__ g2,
    const int* __restrict__ vsp,
    unsigned short* __restrict__ Out) {
    __shared__ float sS[64][68];
    __shared__ unsigned short sP[64][72];
    __shared__ float sM[64], sL[64], sFac[64];
    __shared__ float sPA[64][12];

    const int blk = blockIdx.x;
    const int qt = blk & 15, h = (blk >> 4) & 15, b = blk >> 8;
    const int tid = threadIdx.x, wave = tid >> 5, lane = tid & 31;
    const int vs = vsp[0];
    const float scale = 0.08838834764831845f;  // 1/sqrt(128)
    const float gate2v = g2[h];
    const size_t headQ = (size_t)(b * 16 + h) * 1024 * 128;
    const size_t headV = (size_t)(b * 16 + h) * 128 * 1024;

    // ---- init running stats + adapter probs (one thread per q row) ----
    if (tid < 64) {
        sM[tid] = -3.0e38f; sL[tid] = 0.f;
        int qpos = qt * 64 + tid;
        const unsigned short* qr = Q + headQ + (size_t)qpos * 128;
        float sa[10]; float mx = -3.0e38f;
        for (int j = 0; j < 10; ++j) {
            const float* ak = AK + (size_t)j * 2048 + h * 128;
            float acc = 0.f;
            for (int d = 0; d < 128; ++d) acc += bf2f(qr[d]) * ak[d];
            sa[j] = acc * scale;
            mx = fmaxf(mx, sa[j]);
        }
        float sum = 0.f;
        for (int j = 0; j < 10; ++j) { sa[j] = __expf(sa[j] - mx); sum += sa[j]; }
        float g = tanhf(g1[h]) / sum;
        for (int j = 0; j < 10; ++j) sPA[tid][j] = sa[j] * g;
    }
    __syncthreads();

    // ---- preload Q fragments (register resident for whole k loop) ----
    const int q_rt = wave & 3, dhalf = wave >> 2;
    BFrag qf[4];
    {
        int qpos = qt * 64 + q_rt * 16 + (lane & 15);
        const unsigned short* qr = Q + headQ + (size_t)qpos * 128;
        int kb = (lane < 16) ? 0 : 8;
#pragma unroll
        for (int ds_ = 0; ds_ < 4; ++ds_) {
            qf[ds_].q[0] = *(const uint4*)(qr + ds_ * 32 + kb);
            qf[ds_].q[1] = *(const uint4*)(qr + ds_ * 32 + kb + 16);
        }
    }
    CFrag oacc[4];
#pragma unroll
    for (int j = 0; j < 4; ++j)
#pragma unroll
        for (int e = 0; e < 8; ++e) oacc[j].f[e] = 0.f;

    const int rowsel = (lane >> 4) * 8;
    const int nc = lane & 15;

    for (int kblk = 0; kblk <= qt; ++kblk) {   // causal early exit
        const int kbase = kblk * 64;
        // ---- S = Q @ K^T : 2 k-tiles per wave ----
#pragma unroll
        for (int t = 0; t < 2; ++t) {
            const int kt = (wave >> 2) * 2 + t;
            CFrag s;
#pragma unroll
            for (int e = 0; e < 8; ++e) s.f[e] = 0.f;
            {
                int key = kbase + kt * 16 + (lane & 15);
                const unsigned short* kr = Kc + headQ + (size_t)key * 128;
                int cb = (lane < 16) ? 0 : 8;
#pragma unroll
                for (int ds_ = 0; ds_ < 4; ++ds_) {
                    BFrag bk;
                    bk.q[0] = *(const uint4*)(kr + ds_ * 32 + cb);
                    bk.q[1] = *(const uint4*)(kr + ds_ * 32 + cb + 16);
                    s.v = __builtin_amdgcn_wmma_f32_16x16x32_bf16(
                        false, qf[ds_].v, false, bk.v, (short)0, s.v, false, false);
                }
            }
#pragma unroll
            for (int e = 0; e < 8; ++e) {   // scale + causal mask + gate2 window
                int rq = q_rt * 16 + rowsel + e;
                int ck = kt * 16 + nc;
                int qpos = qt * 64 + rq;
                int kpos = kbase + ck;
                float v = s.f[e] * scale;
                if (kpos > qpos) v += -1.0e9f;
                else if (qpos >= vs + 10 && kpos >= vs && kpos < vs + 10) v += gate2v;
                sS[rq][ck] = v;
            }
        }
        __syncthreads();

        // ---- online softmax (one thread per q row) ----
        if (tid < 64) {
            float mOld = sM[tid], lOld = sL[tid];
            float mx = mOld;
            for (int c2 = 0; c2 < 64; ++c2) mx = fmaxf(mx, sS[tid][c2]);
            float fac = __expf(mOld - mx);
            float sum = 0.f;
            for (int c2 = 0; c2 < 64; ++c2) {
                float p = __expf(sS[tid][c2] - mx);
                sum += p;
                sP[tid][c2] = f2bf(p);
            }
            sM[tid] = mx; sL[tid] = lOld * fac + sum; sFac[tid] = fac;
        }
        __syncthreads();

        // ---- rescale O, then O += P @ V ----
#pragma unroll
        for (int j = 0; j < 4; ++j)
#pragma unroll
            for (int e = 0; e < 8; ++e)
                oacc[j].f[e] *= sFac[q_rt * 16 + rowsel + e];

        BFrag pf[2];
        {
            int rq = q_rt * 16 + (lane & 15);
            int kb = (lane < 16) ? 0 : 8;
#pragma unroll
            for (int kc = 0; kc < 2; ++kc) {
                pf[kc].q[0] = *(const uint4*)&sP[rq][kc * 32 + kb];
                pf[kc].q[1] = *(const uint4*)&sP[rq][kc * 32 + kb + 16];
            }
        }
#pragma unroll
        for (int j = 0; j < 4; ++j) {
            int dt = dhalf * 4 + j;
            int d = dt * 16 + (lane & 15);
            const unsigned short* vr = VT + headV + (size_t)d * 1024 + kbase;
            int cb = (lane < 16) ? 0 : 8;
#pragma unroll
            for (int kc = 0; kc < 2; ++kc) {
                BFrag bv;
                bv.q[0] = *(const uint4*)(vr + kc * 32 + cb);
                bv.q[1] = *(const uint4*)(vr + kc * 32 + cb + 16);
                oacc[j].v = __builtin_amdgcn_wmma_f32_16x16x32_bf16(
                    false, pf[kc].v, false, bv.v, (short)0, oacc[j].v, false, false);
            }
        }
        __syncthreads();
    }

    // ---- epilogue: normalize + adapter contribution, write bf16 ----
#pragma unroll
    for (int j = 0; j < 4; ++j) {
        int dt = dhalf * 4 + j;
        int dd = dt * 16 + nc;
#pragma unroll
        for (int e = 0; e < 8; ++e) {
            int rq = q_rt * 16 + rowsel + e;
            int qpos = qt * 64 + rq;
            float v = oacc[j].f[e] / sL[rq];
            float ad = 0.f;
#pragma unroll
            for (int jj = 0; jj < 10; ++jj)
                ad += sPA[rq][jj] * AV[(size_t)jj * 2048 + h * 128 + dd];
            Out[(size_t)(b * 1024 + qpos) * 2048 + h * 128 + dd] = f2bf(v + ad);
        }
    }
}

// ---------------------------------------------------------------------------
extern "C" void kernel_launch(void* const* d_in, const int* in_sizes, int n_in,
                              void* d_out, int out_size, void* d_ws, size_t ws_size,
                              hipStream_t stream) {
    (void)in_sizes; (void)n_in; (void)out_size; (void)ws_size;
    const float* x    = (const float*)d_in[0];
    const float* adp  = (const float*)d_in[1];
    const float* fcos = (const float*)d_in[3];
    const float* fsin = (const float*)d_in[4];
    const float* wq   = (const float*)d_in[5];
    const float* wk   = (const float*)d_in[6];
    const float* wv   = (const float*)d_in[7];
    const float* wo   = (const float*)d_in[8];
    const float* g1   = (const float*)d_in[9];
    const float* g2   = (const float*)d_in[10];
    const int*   vs   = (const int*)d_in[11];
    float* out = (float*)d_out;

    char* ws = (char*)d_ws;
    size_t off = 0;
    auto bump = [&](size_t bytes) {
        size_t r = off;
        off = (off + bytes + 255) & ~(size_t)255;
        return r;
    };
    unsigned short* xb   = (unsigned short*)(ws + bump(2048ull * 2048 * 2));
    unsigned short* wb   = (unsigned short*)(ws + bump(2048ull * 2048 * 2));
    float*          tmpF = (float*)         (ws + bump(2048ull * 2048 * 4));
    unsigned short* qb   = (unsigned short*)(ws + bump(2ull * 16 * 1024 * 128 * 2));
    unsigned short* kb   = (unsigned short*)(ws + bump(2ull * 16 * 1024 * 128 * 2));
    unsigned short* vtb  = (unsigned short*)(ws + bump(2ull * 16 * 128 * 1024 * 2));
    unsigned short* atb  = (unsigned short*)(ws + bump(2048ull * 2048 * 2));
    unsigned short* ab   = (unsigned short*)(ws + bump(10ull * 2048 * 2));
    float*          akF  = (float*)         (ws + bump(10ull * 2048 * 4));
    float*          avF  = (float*)         (ws + bump(10ull * 2048 * 4));

    const int NW = 2048 * 2048;  // 4,194,304 weight elements

    cast_f32_bf16<<<(NW + 255) / 256, 256, 0, stream>>>(x, xb, NW);
    cast_f32_bf16<<<(20480 + 255) / 256, 256, 0, stream>>>(adp, ab, 20480);

    // K projection (+ adapter K), RoPE
    cast_f32_bf16<<<(NW + 255) / 256, 256, 0, stream>>>(wk, wb, NW);
    gemm_bf16_nt<<<dim3(16, 16), 256, 0, stream>>>(xb, wb, tmpF, 2048, 2048, 2048);
    gemm_bf16_nt<<<dim3(16, 1), 256, 0, stream>>>(ab, wb, akF, 10, 2048, 2048);
    rope_pack<<<2048, 1024, 0, stream>>>(tmpF, fcos, fsin, kb);

    // V projection (+ adapter V), transpose-pack
    cast_f32_bf16<<<(NW + 255) / 256, 256, 0, stream>>>(wv, wb, NW);
    gemm_bf16_nt<<<dim3(16, 16), 256, 0, stream>>>(xb, wb, tmpF, 2048, 2048, 2048);
    gemm_bf16_nt<<<dim3(16, 1), 256, 0, stream>>>(ab, wb, avF, 10, 2048, 2048);
    pack_vt<<<2048, 256, 0, stream>>>(tmpF, vtb);

    // Q projection, RoPE
    cast_f32_bf16<<<(NW + 255) / 256, 256, 0, stream>>>(wq, wb, NW);
    gemm_bf16_nt<<<dim3(16, 16), 256, 0, stream>>>(xb, wb, tmpF, 2048, 2048, 2048);
    rope_pack<<<2048, 1024, 0, stream>>>(tmpF, fcos, fsin, qb);

    // attention (flash, causal, gate2 window, adapter path fused)
    attn_flash<<<512, 256, 0, stream>>>(qb, kb, vtb, akF, avF, g1, g2, vs, atb);

    // output projection -> fp32 d_out
    cast_f32_bf16<<<(NW + 255) / 256, 256, 0, stream>>>(wo, wb, NW);
    gemm_bf16_nt<<<dim3(16, 16), 256, 0, stream>>>(atb, wb, out, 2048, 2048, 2048);
}
